// NewGPTJointAttention_90211493085428
// MI455X (gfx1250) — matrix operand order repaired and checked by reading, not compiled, not run on hardware
//
#include <hip/hip_runtime.h>
#include <hip/hip_bf16.h>

// ---------- problem constants ----------
constexpr int SEQ  = 2048;
constexpr int DIM  = 2048;
constexpr int NH   = 16;
constexpr int HDIM = 128;   // DIM / NH

// ---------- types ----------
typedef __bf16 bf16_t;
typedef __attribute__((ext_vector_type(16))) __bf16 v16bf;
typedef __attribute__((ext_vector_type(8)))  __bf16 v8bf;
typedef __attribute__((ext_vector_type(8)))  float  v8f;

union FragU { v16bf v; v8bf h[2]; };

// Load a 16-element bf16 fragment as two contiguous 16-byte (b128) loads.
__device__ __forceinline__ v16bf ld_frag(const bf16_t* p0, const bf16_t* p1) {
    FragU f;
    f.h[0] = *(const v8bf*)p0;
    f.h[1] = *(const v8bf*)p1;
    return f.v;
}

// =====================================================================
// GEMM:  C[M,N] = A[M,K] * W[N,K]^T   (einsum 'sd,ed->se')
// A is fp32 (stage-1) or bf16 (out-proj); W is fp32. bf16 WMMA, fp32 acc.
// Block: 256 threads = 8 waves, tile 128(M) x 128(N), K-chunk 32.
// Waves arranged 2(M) x 4(N); each wave owns a 64x32 sub-tile =
// 4x2 WMMA tiles -> 8 WMMAs per 6 fragment loads per K-iteration.
// =====================================================================
constexpr int BM = 128, BN = 128, BK = 32;
constexpr int LDSA = 40;   // padded row stride (bf16 elems) -> conflict-free

template <bool ABF16>
__global__ __launch_bounds__(256)
void gemm_nt_kernel(const void* __restrict__ Ap,
                    const float* __restrict__ W,
                    float* __restrict__ C,
                    int M, int N, int K) {
    __shared__ alignas(16) bf16_t sA[BM][LDSA];
    __shared__ alignas(16) bf16_t sW[BN][LDSA];

    const int t    = threadIdx.x;
    const int lane = t & 31;
    const int w    = t >> 5;      // 0..7
    const int wm   = w & 1;       // 0..1  -> M offset wm*64
    const int wn   = w >> 1;      // 0..3  -> N offset wn*32
    const int bm0  = blockIdx.y * BM;
    const int bn0  = blockIdx.x * BN;
    const int l15  = lane & 15;
    const int hi   = lane >> 4;   // 0 or 1

    v8f acc[4][2] = {};

    for (int ko = 0; ko < K; ko += BK) {
        // ---- stage A tile (convert fp32 -> bf16 if needed) ----
        if (ABF16) {
            const bf16_t* A = (const bf16_t*)Ap;
#pragma unroll
            for (int i = 0; i < 2; ++i) {
                int idx = t + 256 * i;          // 0..511, 4 v8bf per row
                int r = idx >> 2;
                int c = (idx & 3) * 8;
                *(v8bf*)&sA[r][c] = *(const v8bf*)&A[(size_t)(bm0 + r) * K + ko + c];
            }
        } else {
            const float* A = (const float*)Ap;
#pragma unroll
            for (int i = 0; i < 4; ++i) {
                int idx = t + 256 * i;          // 0..1023, 8 float4 per row
                int r = idx >> 3;
                int c = (idx & 7) * 4;
                const float4 f = *(const float4*)&A[(size_t)(bm0 + r) * K + ko + c];
                sA[r][c + 0] = (bf16_t)f.x; sA[r][c + 1] = (bf16_t)f.y;
                sA[r][c + 2] = (bf16_t)f.z; sA[r][c + 3] = (bf16_t)f.w;
            }
        }
        // ---- stage W tile: 128x32 fp32 ----
#pragma unroll
        for (int i = 0; i < 4; ++i) {
            int idx = t + 256 * i;              // 0..1023
            int r = idx >> 3;
            int c = (idx & 7) * 4;
            const float4 f = *(const float4*)&W[(size_t)(bn0 + r) * K + ko + c];
            sW[r][c + 0] = (bf16_t)f.x; sW[r][c + 1] = (bf16_t)f.y;
            sW[r][c + 2] = (bf16_t)f.z; sW[r][c + 3] = (bf16_t)f.w;
        }
        if (ko + BK < K) {  // hint next tiles into cache (global_prefetch_b8)
            __builtin_prefetch(&W[(size_t)(bn0 + (t >> 1)) * K + ko + BK], 0, 1);
        }
        __syncthreads();

        // ---- fragments: 4 A-frags + 2 B-frags -> 8 WMMAs ----
        v16bf afr[4], bfr[2];
        const int kb2 = hi ? 8 : 0;    // A-layout: lanes 16-31 start at K=8
        const int kbB = hi ? 8 : 0;    // B-layout: lanes 16-31 start at K=8
#pragma unroll
        for (int mt = 0; mt < 4; ++mt) {
            const bf16_t* row = &sA[wm * 64 + mt * 16 + l15][0];
            afr[mt] = ld_frag(row + kb2, row + kb2 + 16);
        }
#pragma unroll
        for (int nt = 0; nt < 2; ++nt) {
            const bf16_t* row = &sW[wn * 32 + nt * 16 + l15][0];
            bfr[nt] = ld_frag(row + kbB, row + kbB + 16);
        }
#pragma unroll
        for (int mt = 0; mt < 4; ++mt)
#pragma unroll
            for (int nt = 0; nt < 2; ++nt)
                acc[mt][nt] = __builtin_amdgcn_wmma_f32_16x16x32_bf16(
                    false, afr[mt], false, bfr[nt], (short)0, acc[mt][nt], false, false);
        __syncthreads();
    }

    // ---- epilogue: C/D layout row = v + 8*hi, col = l15 ----
#pragma unroll
    for (int mt = 0; mt < 4; ++mt)
#pragma unroll
        for (int nt = 0; nt < 2; ++nt) {
            const int n = bn0 + wn * 32 + nt * 16 + l15;
#pragma unroll
            for (int v = 0; v < 8; ++v) {
                const int m = bm0 + wm * 64 + mt * 16 + v + 8 * hi;
                C[(size_t)m * N + n] = acc[mt][nt][v];
            }
        }
}

// =====================================================================
// Rotary (GPT-J interleaved) + bf16 head-pack:  src fp32 [S,D] ->
// dst bf16 [H][S][HD].  mode: 1 = apply rotary (Q/K), 0 = plain (V).
// out[2i]   = x[2i]*cos - x[2i+1]*sin ; out[2i+1] = x[2i+1]*cos + x[2i]*sin
// =====================================================================
__global__ __launch_bounds__(256)
void rope_pack_kernel(const float* __restrict__ src,
                      const float* __restrict__ sinb,
                      const float* __restrict__ cosb,
                      bf16_t* __restrict__ dst, int do_rope) {
    int idx = blockIdx.x * blockDim.x + threadIdx.x;   // one (even,odd) pair
    if (idx >= SEQ * DIM / 2) return;
    int s = idx / (DIM / 2);
    int j = idx - s * (DIM / 2);
    int e = 2 * j;
    int head = e >> 7;          // / HDIM
    int hd   = e & (HDIM - 1);
    float x0 = src[(size_t)s * DIM + e];
    float x1 = src[(size_t)s * DIM + e + 1];
    float r0 = x0, r1 = x1;
    if (do_rope) {
        int i = hd >> 1;                        // 0..63, sin shape (S, HD/2)
        float c  = cosb[s * (HDIM / 2) + i];
        float sn = sinb[s * (HDIM / 2) + i];
        r0 = x0 * c - x1 * sn;
        r1 = x1 * c + x0 * sn;
    }
    size_t o = ((size_t)head * SEQ + s) * HDIM + hd;
    dst[o]     = (bf16_t)r0;
    dst[o + 1] = (bf16_t)r1;
}

// =====================================================================
// Flash attention (causal), bf16 WMMA QK^T and PV, fp32 online softmax.
// Grid: (SEQ/64, NH), block 128 = 4 waves; wave w owns query rows
// q0 + 16w .. +15 with Q resident in registers (4 A-frags).
// Key blocks of 32 staged in LDS (K row-major, V transposed).
// =====================================================================
__global__ __launch_bounds__(128)
void attn_kernel(const bf16_t* __restrict__ Qh,
                 const bf16_t* __restrict__ Kh,
                 const bf16_t* __restrict__ Vh,
                 bf16_t* __restrict__ Obf) {
    __shared__ alignas(16) bf16_t sK[32][136];    // 32 keys x 128 d (+pad)
    __shared__ alignas(16) bf16_t sVt[128][40];   // transposed: [d][k] (+pad)
    __shared__ alignas(16) bf16_t sP[4][16][40];  // per-wave P tile 16x32 (+pad)

    const int t    = threadIdx.x;
    const int lane = t & 31;
    const int w    = t >> 5;                  // 0..3
    const int head = blockIdx.y;
    const int q0   = blockIdx.x * 64;
    const int l15  = lane & 15;
    const int hi   = lane >> 4;

    const bf16_t* Qb = Qh + (size_t)head * SEQ * HDIM;
    const bf16_t* Kb = Kh + (size_t)head * SEQ * HDIM;
    const bf16_t* Vb = Vh + (size_t)head * SEQ * HDIM;

    // preload this wave's 16x128 Q strip as 4 A-fragments
    v16bf qf[4];
    {
        const int kb2 = hi ? 8 : 0;
        const bf16_t* qrow = Qb + (size_t)(q0 + w * 16 + l15) * HDIM;
#pragma unroll
        for (int dc = 0; dc < 4; ++dc)
            qf[dc] = ld_frag(qrow + dc * 32 + kb2, qrow + dc * 32 + kb2 + 16);
    }

    v8f oacc[8] = {};
    float mrow[8], lrow[8];
#pragma unroll
    for (int v = 0; v < 8; ++v) { mrow[v] = -1e30f; lrow[v] = 0.0f; }

    const int nkb = (q0 + 64) / 32;          // causal: keys 0 .. q0+63
    const float scale = 0.08838834764831845f; // 1/sqrt(HDIM)

    for (int kb = 0; kb < nkb; ++kb) {
        __syncthreads();
        // ---- cooperative stage: K block (row-major) + V block (transposed)
#pragma unroll
        for (int i = 0; i < 4; ++i) {
            int idx = t + 128 * i;           // 0..511 (16 v8bf per key row)
            int r  = idx >> 4;               // 0..31
            int c8 = (idx & 15) * 8;         // 0..120
            const v8bf kv = *(const v8bf*)&Kb[(size_t)(kb * 32 + r) * HDIM + c8];
            *(v8bf*)&sK[r][c8] = kv;
            const v8bf vv = *(const v8bf*)&Vb[(size_t)(kb * 32 + r) * HDIM + c8];
#pragma unroll
            for (int jj = 0; jj < 8; ++jj) sVt[c8 + jj][r] = vv[jj];
        }
        __syncthreads();

        // ---- scores: 2 subtiles x 4 d-chunks = 8 WMMAs ----
        v8f sc[2] = {};
        const int kbB = hi ? 8 : 0;
#pragma unroll
        for (int sub = 0; sub < 2; ++sub) {
            const bf16_t* krow = &sK[sub * 16 + l15][0];
#pragma unroll
            for (int dc = 0; dc < 4; ++dc) {
                v16bf bfr = ld_frag(krow + dc * 32 + kbB, krow + dc * 32 + kbB + 16);
                sc[sub] = __builtin_amdgcn_wmma_f32_16x16x32_bf16(
                    false, qf[dc], false, bfr, (short)0, sc[sub], false, false);
            }
        }

        // ---- scale + causal mask + online softmax ----
        const int kcol = kb * 32 + l15;
        float pmax[8], psum[8];
#pragma unroll
        for (int v = 0; v < 8; ++v) {
            const int qg = q0 + w * 16 + v + 8 * hi;
            float s0 = sc[0][v] * scale;
            float s1 = sc[1][v] * scale;
            if (kcol > qg)      s0 = -1e30f;
            if (kcol + 16 > qg) s1 = -1e30f;
            sc[0][v] = s0; sc[1][v] = s1;
            pmax[v] = fmaxf(s0, s1);
        }
#pragma unroll
        for (int off = 1; off < 16; off <<= 1)
#pragma unroll
            for (int v = 0; v < 8; ++v)
                pmax[v] = fmaxf(pmax[v], __shfl_xor(pmax[v], off, 32));

#pragma unroll
        for (int v = 0; v < 8; ++v) {
            float mn = fmaxf(mrow[v], pmax[v]);
            float sf = __expf(mrow[v] - mn);
            mrow[v] = mn;
            lrow[v] *= sf;
#pragma unroll
            for (int dt = 0; dt < 8; ++dt) oacc[dt][v] *= sf;
            float p0 = __expf(sc[0][v] - mn);
            float p1 = __expf(sc[1][v] - mn);
            psum[v] = p0 + p1;
            // C-layout -> row-major LDS (to re-shape into A-layout)
            sP[w][v + 8 * hi][l15]      = (bf16_t)p0;
            sP[w][v + 8 * hi][16 + l15] = (bf16_t)p1;
        }
#pragma unroll
        for (int off = 1; off < 16; off <<= 1)
#pragma unroll
            for (int v = 0; v < 8; ++v)
                psum[v] += __shfl_xor(psum[v], off, 32);
#pragma unroll
        for (int v = 0; v < 8; ++v) lrow[v] += psum[v];

        // ---- P (16x32) x V (32x128): 8 WMMAs accumulating O ----
        v16bf pf;
        {
            const int kb2 = hi ? 8 : 0;
            const bf16_t* prow = &sP[w][l15][0];
            pf = ld_frag(prow + kb2, prow + kb2 + 16);
        }
#pragma unroll
        for (int dt = 0; dt < 8; ++dt) {
            const bf16_t* vrow = &sVt[dt * 16 + l15][0];
            v16bf bfr = ld_frag(vrow + kbB, vrow + kbB + 16);
            oacc[dt] = __builtin_amdgcn_wmma_f32_16x16x32_bf16(
                false, pf, false, bfr, (short)0, oacc[dt], false, false);
        }
    }

    // ---- normalize + merge heads: Obf[s][head*128 + d] (bf16) ----
#pragma unroll
    for (int v = 0; v < 8; ++v) {
        float inv = 1.0f / lrow[v];
        const int m = q0 + w * 16 + v + 8 * hi;
#pragma unroll
        for (int dt = 0; dt < 8; ++dt)
            Obf[(size_t)m * DIM + head * HDIM + dt * 16 + l15] =
                (bf16_t)(oacc[dt][v] * inv);
    }
}

// =====================================================================
// launcher
// =====================================================================
extern "C" void kernel_launch(void* const* d_in, const int* in_sizes, int n_in,
                              void* d_out, int out_size, void* d_ws, size_t ws_size,
                              hipStream_t stream) {
    (void)in_sizes; (void)n_in; (void)out_size; (void)ws_size;
    const float* X    = (const float*)d_in[0];   // hidden_states [1,S,D]
    // d_in[1] attention_mask: implicit causal, unused
    const float* sinb = (const float*)d_in[2];   // [S, HD/2]
    const float* cosb = (const float*)d_in[3];   // [S, HD/2]
    const float* Wq   = (const float*)d_in[4];
    const float* Wk   = (const float*)d_in[5];
    const float* Wv   = (const float*)d_in[6];
    const float* Wo   = (const float*)d_in[7];

    char* ws = (char*)d_ws;
    float*  stagef = (float*)ws;                              // 16 MB fp32 stage
    bf16_t* Qh  = (bf16_t*)(ws + (16u << 20));                //  8 MB
    bf16_t* Kh  = (bf16_t*)(ws + (24u << 20));                //  8 MB
    bf16_t* Vh  = (bf16_t*)(ws + (32u << 20));                //  8 MB
    bf16_t* Obf = (bf16_t*)(ws + (40u << 20));                //  8 MB (peak 48 MB)

    dim3 ggrid(DIM / BN, SEQ / BM);   // (16, 16)

    const float* Ws[3] = {Wq, Wk, Wv};
    bf16_t* dsts[3] = {Qh, Kh, Vh};
    for (int m = 0; m < 3; ++m) {
        gemm_nt_kernel<false><<<ggrid, 256, 0, stream>>>(X, Ws[m], stagef,
                                                         SEQ, DIM, DIM);
        int npair = SEQ * DIM / 2;
        rope_pack_kernel<<<(npair + 255) / 256, 256, 0, stream>>>(
            stagef, sinb, cosb, dsts[m], (m < 2) ? 1 : 0);
    }

    attn_kernel<<<dim3(SEQ / 64, NH), 128, 0, stream>>>(Qh, Kh, Vh, Obf);

    gemm_nt_kernel<true><<<ggrid, 256, 0, stream>>>(Obf, Wo, (float*)d_out,
                                                    SEQ, DIM, DIM);
}
